// GINWithCausalAttention_75514114998660
// MI455X (gfx1250) — matrix operand order repaired and checked by reading, not compiled, not run on hardware
//
#include <hip/hip_runtime.h>
#include <hip/hip_bf16.h>

// ---------------------------------------------------------------------------
// MI455X (gfx1250) implementation of GIN + causal attention forward.
//
// Roofline: 7x [50000x128]@[128x128] GEMMs ~= 11.5 GFLOP over ~25.6MB hot
// activations (L2-resident on 192MB L2) -> compute-limited -> use
// v_wmma_f32_16x16x32_f16 (f16 A/B, f32 accum). Edge scatter (102M f32
// atomic adds) and pooling are HBM/L2-bandwidth pieces done with atomics.
// Weights are pre-packed to f16 W^T and staged in LDS (<=32KB of 320KB/WGP).
// ---------------------------------------------------------------------------

typedef __attribute__((ext_vector_type(16))) _Float16 v16h;
typedef __attribute__((ext_vector_type(8)))  _Float16 v8h;
typedef __attribute__((ext_vector_type(8)))  float    v8f;

union Frag {
  v16h v;
  struct { v8h lo; v8h hi; } h;
};

#define GIN_N 50000
#define GIN_E 800000
#define GIN_D 128
#define GIN_B 512
#define GIN_L 3

// ---------------------------------------------------------------------------
// Generic WMMA GEMM: out[M x (NT*16)] = epilogue(A[M x K] @ WT^T), A,WT in f16.
// WT is stored transposed/row-major as [NT*16][K] (pre-packed f16).
// One block = 8 waves = 128 rows; wave w -> rows [row0, row0+16).
// A frag layout (16-bit 16x32, ISA 7.12.2): lane m=lane&15, kbase=(lane>>4)*8,
//   VGPR0..3 -> K=kbase..kbase+7, VGPR4..7 -> K=16+kbase..16+kbase+7.
// B frag layout (same K order as A): lane n=lane&15, kbase=(lane>>4)*8,
//   VGPR0..3 -> K=kbase..kbase+7, VGPR4..7 -> K=16+kbase..16+kbase+7.
// C/D layout: VGPR r -> row r + 8*(lane>>4), col = lane&15.
// mode: 0 = relu(s0*acc+b0)
//       1 = relu(s1*relu(s0*acc+b0)+b1)      (inner BN + outer BN fused)
//       2 = leaky_relu_0.2(s0*acc+b0)
// ---------------------------------------------------------------------------
template<int NT>
__global__ __launch_bounds__(256) void wmma_gemm(
    const _Float16* __restrict__ A, const _Float16* __restrict__ WT,
    const float* __restrict__ s0, const float* __restrict__ b0,
    const float* __restrict__ s1, const float* __restrict__ b1,
    int M, int K, int mode,
    float* __restrict__ outF, _Float16* __restrict__ outH)
{
  extern __shared__ char smem_raw[];
  _Float16* lds = (_Float16*)smem_raw;
  const int Nc = NT * 16;

  // Stage the f16 W^T tile into LDS (broadcast to all 8 waves).
  {
    const float4* sg = (const float4*)WT;
    float4* dg = (float4*)smem_raw;
    int nvec = (Nc * K) / 8;   // 8 halves per float4
    for (int j = threadIdx.x; j < nvec; j += 256) dg[j] = sg[j];
  }
  __syncthreads();

  const int wave  = threadIdx.x >> 5;
  const int lane  = threadIdx.x & 31;
  const int row0  = blockIdx.x * 128 + wave * 16;
  if (row0 >= M) return;          // M is a multiple of 16 in every call site

  const int lhalf = lane >> 4;
  const int lq    = lane & 15;

  v8f acc[NT] = {};

  const _Float16* arow = A + (size_t)(row0 + lq) * K;

  for (int k0 = 0; k0 < K; k0 += 32) {
    Frag af;
    af.h.lo = *(const v8h*)(arow + k0 + lhalf * 8);
    af.h.hi = *(const v8h*)(arow + k0 + 16 + lhalf * 8);
#pragma unroll
    for (int nt = 0; nt < NT; ++nt) {
      Frag bf;
      const _Float16* bp = lds + (size_t)(nt * 16 + lq) * K + k0 + lhalf * 8;
      bf.h.lo = *(const v8h*)(bp);
      bf.h.hi = *(const v8h*)(bp + 16);
      acc[nt] = __builtin_amdgcn_wmma_f32_16x16x32_f16(
          /*neg_a=*/false, af.v, /*neg_b=*/false, bf.v,
          /*c_mod=*/(short)0, acc[nt], /*reuse_a=*/false, /*reuse_b=*/false);
    }
  }

  // Fused epilogue.
#pragma unroll
  for (int nt = 0; nt < NT; ++nt) {
    const int col = nt * 16 + lq;
    const float sc0 = s0[col], bc0 = b0[col];
    float sc1 = 0.f, bc1 = 0.f;
    if (mode == 1) { sc1 = s1[col]; bc1 = b1[col]; }
#pragma unroll
    for (int r = 0; r < 8; ++r) {
      const int row = row0 + r + 8 * lhalf;
      float v = acc[nt][r];
      float t = sc0 * v + bc0;
      if (mode == 2)      v = (t > 0.f) ? t : 0.2f * t;
      else                v = fmaxf(t, 0.f);
      if (mode == 1)      v = fmaxf(sc1 * v + bc1, 0.f);
      const size_t o = (size_t)row * Nc + col;
      if (outF) outF[o] = v;
      if (outH) outH[o] = (_Float16)v;
    }
  }
}

// --------------------------- helper kernels --------------------------------

__global__ void zero_f32(float* __restrict__ p, size_t n) {
  size_t i = (size_t)blockIdx.x * blockDim.x + threadIdx.x;
  if (i < n) p[i] = 0.f;
}

__global__ void f32_to_f16(const float* __restrict__ a, _Float16* __restrict__ b, size_t n) {
  size_t i = (size_t)blockIdx.x * blockDim.x + threadIdx.x;
  if (i < n) b[i] = (_Float16)a[i];
}

// Transpose + f32->f16 weight pack: W[K x Nc] -> WT[Nc x K] f16.
__global__ void pack_wt(const float* __restrict__ W, _Float16* __restrict__ WT, int K, int Nc) {
  int i = blockIdx.x * blockDim.x + threadIdx.x;
  if (i >= K * Nc) return;
  int n = i / K, k = i % K;
  WT[i] = (_Float16)W[k * Nc + n];
}

__global__ void init_ones(float* __restrict__ p) { p[threadIdx.x] = 1.0f; }

// Fold BatchNorm (eval mode) + linear bias into per-column affines.
__global__ void affine_prep(
    const float* b1, const float* g1, const float* be1, const float* rm1, const float* rv1,
    const float* b2, const float* g2, const float* be2, const float* rm2, const float* rv2,
    const float* og, const float* ob, const float* orm, const float* orv,
    float* s1, float* t1, float* s2, float* t2, float* so, float* to)
{
  const int idx = blockIdx.x * GIN_D + threadIdx.x;
  float sa = g1[idx] * rsqrtf(rv1[idx] + 1e-5f);
  s1[idx] = sa;
  t1[idx] = (b1[idx] - rm1[idx]) * sa + be1[idx];
  float sb = g2[idx] * rsqrtf(rv2[idx] + 1e-5f);
  s2[idx] = sb;
  t2[idx] = (b2[idx] - rm2[idx]) * sb + be2[idx];
  float sc = og[idx] * rsqrtf(orv[idx] + 1e-5f);
  so[idx] = sc;
  to[idx] = ob[idx] - orm[idx] * sc;
}

// GIN aggregation: agg[dst] += h[src] over edges. One wave per edge, float4/lane.
__global__ void scatter_add(const int* __restrict__ src, const int* __restrict__ dst,
                            const float* __restrict__ h, float* __restrict__ agg, int E) {
  int t = blockIdx.x * blockDim.x + threadIdx.x;
  int e = t >> 5;
  if (e >= E) return;
  int lane = t & 31;
  const float4 v = *(const float4*)(h + (size_t)src[e] * GIN_D + lane * 4);
  float* a = agg + (size_t)dst[e] * GIN_D + lane * 4;
  atomicAdd(a + 0, v.x);
  atomicAdd(a + 1, v.y);
  atomicAdd(a + 2, v.z);
  atomicAdd(a + 3, v.w);
}

// m = (1+eps)*h + agg, converted to f16 for the next WMMA GEMM.
__global__ void combine_eps(const float* __restrict__ h, const float* __restrict__ agg,
                            const float* __restrict__ eps, int li,
                            _Float16* __restrict__ out, size_t n) {
  size_t i = (size_t)blockIdx.x * blockDim.x + threadIdx.x;
  if (i >= n) return;
  float e = 1.0f + eps[li];
  out[i] = (_Float16)(e * h[i] + agg[i]);
}

// Attention head (32->1), sigmoid score, and attention-weighted segment-sum
// pooling. One wave per node; wave32 xor-shuffle reduction for the dot.
__global__ void score_pool(const _Float16* __restrict__ a2, const float* __restrict__ aW3,
                           const float* __restrict__ ab3, const float* __restrict__ temp,
                           const float* __restrict__ hF, const int* __restrict__ batch,
                           float* __restrict__ ge, float* __restrict__ scores, int Nn) {
  int t = blockIdx.x * blockDim.x + threadIdx.x;
  int node = t >> 5;
  if (node >= Nn) return;
  int lane = t & 31;
  float v = (float)a2[(size_t)node * 32 + lane] * aW3[lane];
#pragma unroll
  for (int off = 16; off > 0; off >>= 1) v += __shfl_xor(v, off, 32);
  float logit = v + ab3[0];
  logit = fminf(10.f, fmaxf(-10.f, logit));
  float s = 1.f / (1.f + __expf(-logit * temp[0]));
  if (lane == 0) scores[node] = s;
  int b = batch[node];
  const float* hr = hF + (size_t)node * GIN_D;
  float* gr = ge + (size_t)b * GIN_D;
#pragma unroll
  for (int j = 0; j < 4; ++j) {
    int d = lane + 32 * j;
    atomicAdd(&gr[d], hr[d] * s);
  }
}

// Final classifier layer: logits[b,j] = c1[b,:] . cW2[:,j] + cb2[j]  (Ncols=3).
__global__ void classifier2(const float* __restrict__ c1, const float* __restrict__ cW2,
                            const float* __restrict__ cb2, float* __restrict__ logits, int Bn) {
  int t = blockIdx.x * blockDim.x + threadIdx.x;
  if (t >= Bn * 3) return;
  int b = t / 3, j = t % 3;
  float s = cb2[j];
  const float* row = c1 + (size_t)b * GIN_D;
  for (int k = 0; k < GIN_D; ++k) s += row[k] * cW2[k * 3 + j];
  logits[t] = s;
}

// ---------------------------------------------------------------------------

extern "C" void kernel_launch(void* const* d_in, const int* in_sizes, int n_in,
                              void* d_out, int out_size, void* d_ws, size_t ws_size,
                              hipStream_t stream) {
  (void)in_sizes; (void)n_in; (void)out_size; (void)ws_size;

  const float* x    = (const float*)d_in[0];
  const int*   eidx = (const int*)d_in[1];
  const int*   batch= (const int*)d_in[2];
  const float* encW = (const float*)d_in[3];
  const float* encB = (const float*)d_in[4];
  const float* W1   = (const float*)d_in[5];
  const float* b1   = (const float*)d_in[6];
  const float* g1   = (const float*)d_in[7];
  const float* be1  = (const float*)d_in[8];
  const float* rm1  = (const float*)d_in[9];
  const float* rv1  = (const float*)d_in[10];
  const float* W2   = (const float*)d_in[11];
  const float* b2   = (const float*)d_in[12];
  const float* g2   = (const float*)d_in[13];
  const float* be2  = (const float*)d_in[14];
  const float* rm2  = (const float*)d_in[15];
  const float* rv2  = (const float*)d_in[16];
  const float* eps  = (const float*)d_in[17];
  const float* og   = (const float*)d_in[18];
  const float* ob   = (const float*)d_in[19];
  const float* orm  = (const float*)d_in[20];
  const float* orv  = (const float*)d_in[21];
  const float* aW1  = (const float*)d_in[22];
  const float* ab1  = (const float*)d_in[23];
  const float* aW2  = (const float*)d_in[24];
  const float* ab2  = (const float*)d_in[25];
  const float* aW3  = (const float*)d_in[26];
  const float* ab3  = (const float*)d_in[27];
  const float* temp = (const float*)d_in[28];
  const float* cW1  = (const float*)d_in[29];
  const float* cb1  = (const float*)d_in[30];
  const float* cW2  = (const float*)d_in[31];
  const float* cb2  = (const float*)d_in[32];

  const int N = GIN_N, E = GIN_E, D = GIN_D, B = GIN_B, L = GIN_L;
  const size_t ND = (size_t)N * D;

  // d_out layout: logits[B*3] | graph_embed[B*D] | scores[N]
  float* outLogits = (float*)d_out;
  float* outGE     = outLogits + B * 3;
  float* outScores = outGE + (size_t)B * D;

  // ---- bump allocator over d_ws ----
  char* ws = (char*)d_ws;
  size_t off = 0;
  auto alloc = [&](size_t bytes) -> void* {
    void* p = ws + off;
    off = (off + bytes + 255) & ~(size_t)255;
    return p;
  };

  _Float16* wtEnc = (_Float16*)alloc((size_t)D * D * 2);
  _Float16* wtW1[GIN_L]; _Float16* wtW2[GIN_L];
  for (int i = 0; i < L; ++i) wtW1[i] = (_Float16*)alloc((size_t)D * D * 2);
  for (int i = 0; i < L; ++i) wtW2[i] = (_Float16*)alloc((size_t)D * D * 2);
  _Float16* wtC1 = (_Float16*)alloc((size_t)D * D * 2);
  _Float16* wtA1 = (_Float16*)alloc((size_t)64 * D * 2);
  _Float16* wtA2 = (_Float16*)alloc((size_t)32 * 64 * 2);
  float* ones = (float*)alloc(D * 4);
  float* s1 = (float*)alloc((size_t)L * D * 4);
  float* t1 = (float*)alloc((size_t)L * D * 4);
  float* s2 = (float*)alloc((size_t)L * D * 4);
  float* t2 = (float*)alloc((size_t)L * D * 4);
  float* so = (float*)alloc((size_t)L * D * 4);
  float* to = (float*)alloc((size_t)L * D * 4);
  float*    hF  = (float*)alloc(ND * 4);
  float*    agg = (float*)alloc(ND * 4);
  _Float16* hH  = (_Float16*)alloc(ND * 2);
  _Float16* mH  = (_Float16*)alloc(ND * 2);   // also hosts x_f16 for the encoder
  _Float16* tH  = (_Float16*)alloc(ND * 2);
  _Float16* a1H = (_Float16*)alloc((size_t)N * 64 * 2);
  _Float16* a2H = (_Float16*)alloc((size_t)N * 32 * 2);
  _Float16* geH = (_Float16*)alloc((size_t)B * D * 2);
  float*    c1F = (float*)alloc((size_t)B * D * 4);

  auto blocks = [](size_t n) { return (unsigned)((n + 255) / 256); };

  // ---- weight packing (f32 -> f16 transposed) ----
  pack_wt<<<blocks((size_t)D * D), 256, 0, stream>>>(encW, wtEnc, D, D);
  for (int i = 0; i < L; ++i) {
    pack_wt<<<blocks((size_t)D * D), 256, 0, stream>>>(W1 + (size_t)i * D * D, wtW1[i], D, D);
    pack_wt<<<blocks((size_t)D * D), 256, 0, stream>>>(W2 + (size_t)i * D * D, wtW2[i], D, D);
  }
  pack_wt<<<blocks((size_t)D * D), 256, 0, stream>>>(cW1, wtC1, D, D);
  pack_wt<<<blocks((size_t)D * 64), 256, 0, stream>>>(aW1, wtA1, D, 64);
  pack_wt<<<blocks((size_t)64 * 32), 256, 0, stream>>>(aW2, wtA2, 64, 32);
  init_ones<<<1, D, 0, stream>>>(ones);
  affine_prep<<<L, D, 0, stream>>>(b1, g1, be1, rm1, rv1, b2, g2, be2, rm2, rv2,
                                   og, ob, orm, orv, s1, t1, s2, t2, so, to);

  // ---- encoder: h = relu(x @ enc_W + enc_b) ----
  f32_to_f16<<<blocks(ND), 256, 0, stream>>>(x, mH, ND);
  wmma_gemm<8><<<(N + 127) / 128, 256, 8 * 16 * D * 2, stream>>>(
      mH, wtEnc, ones, encB, nullptr, nullptr, N, D, /*mode=*/0, hF, hH);

  // ---- GIN layers ----
  const int* esrc = eidx;
  const int* edst = eidx + E;
  for (int i = 0; i < L; ++i) {
    zero_f32<<<blocks(ND), 256, 0, stream>>>(agg, ND);
    scatter_add<<<blocks((size_t)E * 32), 256, 0, stream>>>(esrc, edst, hF, agg, E);
    combine_eps<<<blocks(ND), 256, 0, stream>>>(hF, agg, eps, i, mH, ND);
    // m = relu(bn1(m @ W1 + b1))
    wmma_gemm<8><<<(N + 127) / 128, 256, 8 * 16 * D * 2, stream>>>(
        mH, wtW1[i], s1 + i * D, t1 + i * D, nullptr, nullptr, N, D, /*mode=*/0,
        nullptr, tH);
    // h = relu(bn_outer(relu(bn2(m @ W2 + b2))))
    wmma_gemm<8><<<(N + 127) / 128, 256, 8 * 16 * D * 2, stream>>>(
        tH, wtW2[i], s2 + i * D, t2 + i * D, so + i * D, to + i * D, N, D, /*mode=*/1,
        hF, hH);
  }

  // ---- attention MLP ----
  wmma_gemm<4><<<(N + 127) / 128, 256, 4 * 16 * D * 2, stream>>>(
      hH, wtA1, ones, ab1, nullptr, nullptr, N, D, /*mode=*/2, nullptr, a1H);
  wmma_gemm<2><<<(N + 127) / 128, 256, 2 * 16 * 64 * 2, stream>>>(
      a1H, wtA2, ones, ab2, nullptr, nullptr, N, 64, /*mode=*/2, nullptr, a2H);

  // ---- scores + attention-weighted global add pool ----
  zero_f32<<<blocks((size_t)B * D), 256, 0, stream>>>(outGE, (size_t)B * D);
  score_pool<<<blocks((size_t)N * 32), 256, 0, stream>>>(
      a2H, aW3, ab3, temp, hF, batch, outGE, outScores, N);

  // ---- classifier ----
  f32_to_f16<<<blocks((size_t)B * D), 256, 0, stream>>>(outGE, geH, (size_t)B * D);
  wmma_gemm<8><<<(B + 127) / 128, 256, 8 * 16 * D * 2, stream>>>(
      geH, wtC1, ones, cb1, nullptr, nullptr, B, D, /*mode=*/0, c1F, nullptr);
  classifier2<<<blocks((size_t)B * 3), 256, 0, stream>>>(c1F, cW2, cb2, outLogits, B);
}
